// LSTMModel_29850022707256
// MI455X (gfx1250) — matrix operand order repaired and checked by reading, not compiled, not run on hardware
//
#include <hip/hip_runtime.h>

#define TLEN  512
#define BATCH 1024
#define DIN   32
#define HID   64

typedef __attribute__((ext_vector_type(16))) __bf16 v16bf;
typedef __attribute__((ext_vector_type(8)))  float  v8f;

#if __has_builtin(__builtin_amdgcn_exp2f)
#define EXP2F __builtin_amdgcn_exp2f
#else
#define EXP2F exp2f
#endif
#if __has_builtin(__builtin_amdgcn_rcpf)
#define RCPF __builtin_amdgcn_rcpf
#else
#define RCPF(x) (1.0f / (x))
#endif

__device__ __forceinline__ float fsig(float x) {
  // sigmoid(x) = 1 / (1 + 2^(-x*log2(e)))
  return RCPF(1.0f + EXP2F(-1.4426950408889634f * x));
}
__device__ __forceinline__ float ftanh(float x) {
  // tanh(x) = 2*sigmoid(2x) - 1
  return 2.0f * fsig(2.0f * x) - 1.0f;
}

__device__ __forceinline__ v8f wmma_bf16(v16bf a, v16bf b, v8f c) {
  // D = A(16x32 bf16) * B(32x16 bf16) + C(16x16 f32)
  return __builtin_amdgcn_wmma_f32_16x16x32_bf16(false, a, false, b, (short)0, c,
                                                 false, false);
}

// 16 contiguous f32 -> bf16 B-fragment lane data (B 32x16: lane=N col,
// lanes 0-15 hold K=0..15, lanes 16-31 hold K=16..31, 2 per VGPR).
__device__ __forceinline__ v16bf cvt16(const float* __restrict__ p) {
  v16bf r;
#pragma unroll
  for (int j = 0; j < 16; ++j) r[j] = (__bf16)p[j];
  return r;
}

// A-fragment (16x32 bf16) from an LDS row-major [16][HID] bf16 tile.
// Lane l: M = l&15; lanes<16 take K = kb..kb+15 with kb=32*kf,
// lanes>=16 the same shifted by 16 (contiguous K, matching B fragments).
__device__ __forceinline__ v16bf a_frag(const __bf16* __restrict__ buf, int lq,
                                        int hi, int kf) {
  const int kb = kf * 32 + (hi ? 16 : 0);
  const uint4* p = reinterpret_cast<const uint4*>(buf + lq * HID + kb);
  union { uint4 q[2]; v16bf v; } u;
  u.q[0] = p[0];   // K = kb .. kb+7
  u.q[1] = p[1];   // K = kb+8 .. kb+15   (+16 bytes)
  return u.v;
}

// A-fragment (16x32) for x_t: row-major f32 [16][DIN] with row stride TLEN*DIN.
__device__ __forceinline__ v16bf x_frag(const float* __restrict__ xr, int hi) {
  const int kb = hi ? 16 : 0;
  v16bf r;
#pragma unroll
  for (int j = 0; j < 8; ++j) {
    r[j]     = (__bf16)xr[kb + j];
    r[j + 8] = (__bf16)xr[kb + 8 + j];
  }
  return r;
}

__global__ __launch_bounds__(256, 1)
void lstm2_fused_wmma(const float* __restrict__ x,
                      const float* __restrict__ Wih0, const float* __restrict__ Whh0,
                      const float* __restrict__ bih0, const float* __restrict__ bhh0,
                      const float* __restrict__ Wih1, const float* __restrict__ Whh1,
                      const float* __restrict__ bih1, const float* __restrict__ bhh1,
                      const float* __restrict__ fc1w, const float* __restrict__ fc1b,
                      const float* __restrict__ fc2w, const float* __restrict__ fc2b,
                      float* __restrict__ out) {
  // Double-buffered bf16 h tiles (row-major [16][HID]) exchanged between the
  // layer-0 wave group (waves 0-3) and the layer-1 wave group (waves 4-7).
  __shared__ __align__(16) __bf16 sHa[2][16 * HID];
  __shared__ __align__(16) __bf16 sHb[2][16 * HID];
  __shared__ __align__(16) float  sHf[16 * HID];  // f32 h2[T-1] for the FC head

  const int tid  = threadIdx.x;
  const int lane = tid & 31;
  const int wave = tid >> 5;
  const int wl   = wave & 3;         // wave index within its layer group
  const int lq   = lane & 15;
  const int hi   = (lane >> 4) & 1;  // lane half (A/B K-offset select)
  const int mb   = hi ? 8 : 0;       // C/D row base for this lane half
  const int b0   = blockIdx.x * 16;  // batch tile origin

  for (int i = tid; i < 16 * HID; i += 256) {
    sHa[0][i] = (__bf16)0.0f; sHa[1][i] = (__bf16)0.0f;
    sHb[0][i] = (__bf16)0.0f; sHb[1][i] = (__bf16)0.0f;
  }
  __syncthreads();

  if (wave < 4) {
    // ---------------- Layer 0: gates = x_t @ Wih0^T + h_a @ Whh0^T + b ----
    // This wave owns h-columns [16*wl, 16*wl+16) => gate N-tiles {4g + wl}.
    v16bf Wi0[4];      // B-fragments of Wih0^T, K=32
    v16bf Wh0[4][2];   // B-fragments of Whh0^T, K=64 (two K-halves)
    float bias0[4];
#pragma unroll
    for (int g = 0; g < 4; ++g) {
      const int n = g * HID + wl * 16 + lq;   // gate column owned by this lane
      Wi0[g]    = cvt16(Wih0 + n * DIN + (hi ? 16 : 0));
      Wh0[g][0] = cvt16(Whh0 + n * HID + (hi ? 16 : 0));
      Wh0[g][1] = cvt16(Whh0 + n * HID + 32 + (hi ? 16 : 0));
      bias0[g]  = bih0[n] + bhh0[n];
    }
    const float* xlane = x + (size_t)(b0 + lq) * TLEN * DIN;  // A row for M=lq
    v8f c0 = (v8f)0.0f;

    for (int k = 0; k <= TLEN; ++k) {
      if (k < TLEN) {
        const __bf16* haR = sHa[(k + 1) & 1];        // h_a[k-1]
        v16bf A0 = a_frag(haR, lq, hi, 0);
        v16bf A1 = a_frag(haR, lq, hi, 1);
        v16bf Xa = x_frag(xlane + k * DIN, hi);
        if (k + 8 < TLEN) __builtin_prefetch(xlane + (k + 8) * DIN, 0, 0);
        v8f gg[4];
#pragma unroll
        for (int g = 0; g < 4; ++g) {
          v8f acc = (v8f)bias0[g];
          acc = wmma_bf16(Xa, Wi0[g], acc);          // K=32 input GEMM
          acc = wmma_bf16(A0, Wh0[g][0], acc);       // K=0..31 recurrent
          acc = wmma_bf16(A1, Wh0[g][1], acc);       // K=32..63 recurrent
          gg[g] = acc;
        }
        __bf16* haW = sHa[k & 1];                    // h_a[k]
#pragma unroll
        for (int e = 0; e < 8; ++e) {
          const float ig = fsig(gg[0][e]);
          const float fg = fsig(gg[1][e]);
          const float gv = ftanh(gg[2][e]);
          const float og = fsig(gg[3][e]);
          const float c  = fg * c0[e] + ig * gv;
          c0[e] = c;
          const float h = og * ftanh(c);
          haW[(mb + e) * HID + wl * 16 + lq] = (__bf16)h;
        }
      }
      __syncthreads();
    }
  } else {
    // ---------------- Layer 1 (lags one step): uses h_a[k-1], h_b[k-2] ----
    v16bf Wi1[4][2];   // B-fragments of Wih1^T, K=64
    v16bf Wh1[4][2];   // B-fragments of Whh1^T, K=64
    float bias1[4];
#pragma unroll
    for (int g = 0; g < 4; ++g) {
      const int n = g * HID + wl * 16 + lq;
      Wi1[g][0] = cvt16(Wih1 + n * HID + (hi ? 16 : 0));
      Wi1[g][1] = cvt16(Wih1 + n * HID + 32 + (hi ? 16 : 0));
      Wh1[g][0] = cvt16(Whh1 + n * HID + (hi ? 16 : 0));
      Wh1[g][1] = cvt16(Whh1 + n * HID + 32 + (hi ? 16 : 0));
      bias1[g]  = bih1[n] + bhh1[n];
    }
    v8f c1 = (v8f)0.0f;

    for (int k = 0; k <= TLEN; ++k) {
      if (k >= 1) {
        const __bf16* haR = sHa[(k - 1) & 1];        // h_a[k-1]
        const __bf16* hbR = sHb[k & 1];              // h_b[k-2]
        v16bf A0 = a_frag(haR, lq, hi, 0);
        v16bf A1 = a_frag(haR, lq, hi, 1);
        v16bf H0 = a_frag(hbR, lq, hi, 0);
        v16bf H1 = a_frag(hbR, lq, hi, 1);
        v8f gg[4];
#pragma unroll
        for (int g = 0; g < 4; ++g) {
          v8f acc = (v8f)bias1[g];
          acc = wmma_bf16(A0, Wi1[g][0], acc);
          acc = wmma_bf16(A1, Wi1[g][1], acc);
          acc = wmma_bf16(H0, Wh1[g][0], acc);
          acc = wmma_bf16(H1, Wh1[g][1], acc);
          gg[g] = acc;
        }
        __bf16* hbW = sHb[(k - 1) & 1];              // h_b[k-1]
        const bool last = (k == TLEN);
#pragma unroll
        for (int e = 0; e < 8; ++e) {
          const float ig = fsig(gg[0][e]);
          const float fg = fsig(gg[1][e]);
          const float gv = ftanh(gg[2][e]);
          const float og = fsig(gg[3][e]);
          const float c  = fg * c1[e] + ig * gv;
          c1[e] = c;
          const float h = og * ftanh(c);
          const int idx = (mb + e) * HID + wl * 16 + lq;
          hbW[idx] = (__bf16)h;
          if (last) sHf[idx] = h;                    // keep f32 copy for FC
        }
      }
      __syncthreads();
    }
  }
  // Last workgroup barrier of both loops makes sHf visible here.

  // ---------------- FC head: out = relu(h2 @ fc1^T + b1) @ fc2^T + b2 ------
  if (wave == 4 && lane < 16) {
    const int r = lane;
    float acc2 = fc2b[0];
    for (int j = 0; j < HID / 2; ++j) {
      float s = fc1b[j];
#pragma unroll 8
      for (int k2 = 0; k2 < HID; ++k2)
        s = fmaf(sHf[r * HID + k2], fc1w[j * HID + k2], s);
      s = fmaxf(s, 0.0f);
      acc2 = fmaf(s, fc2w[j], acc2);
    }
    out[b0 + r] = acc2;
  }
}

extern "C" void kernel_launch(void* const* d_in, const int* in_sizes, int n_in,
                              void* d_out, int out_size, void* d_ws, size_t ws_size,
                              hipStream_t stream) {
  (void)in_sizes; (void)n_in; (void)out_size; (void)d_ws; (void)ws_size;
  const float* x    = (const float*)d_in[0];
  const float* Wih0 = (const float*)d_in[1];
  const float* Whh0 = (const float*)d_in[2];
  const float* bih0 = (const float*)d_in[3];
  const float* bhh0 = (const float*)d_in[4];
  const float* Wih1 = (const float*)d_in[5];
  const float* Whh1 = (const float*)d_in[6];
  const float* bih1 = (const float*)d_in[7];
  const float* bhh1 = (const float*)d_in[8];
  const float* fc1w = (const float*)d_in[9];
  const float* fc1b = (const float*)d_in[10];
  const float* fc2w = (const float*)d_in[11];
  const float* fc2b = (const float*)d_in[12];

  lstm2_fused_wmma<<<dim3(BATCH / 16), dim3(256), 0, stream>>>(
      x, Wih0, Whh0, bih0, bhh0, Wih1, Whh1, bih1, bhh1,
      fc1w, fc1b, fc2w, fc2b, (float*)d_out);
}
